// TemperGraph_65180423684414
// MI455X (gfx1250) — matrix operand order repaired and checked by reading, not compiled, not run on hardware
//
#include <hip/hip_runtime.h>
#include <hip/hip_bf16.h>

// ---------------------------------------------------------------------------
// Problem constants (from reference setup_inputs)
// ---------------------------------------------------------------------------
constexpr int BB  = 12288;   // batch / patches
constexpr int DD  = 1024;    // input dim
constexpr int HH  = 512;     // hidden dim
constexpr int TT  = 12;      // tempers
constexpr int KOP = 3;       // operators per temper
constexpr int GG  = BB / TT; // 1024 patches per temper
constexpr int KP  = 544;     // router K padded: 512+7 -> 544 (multiple of 32)
constexpr int NPL = 16;      // logits N padded: 13 -> 16

typedef __bf16 bf16_t;
typedef __attribute__((ext_vector_type(16))) __bf16 v16bf;
typedef __attribute__((ext_vector_type(8)))  __bf16 v8bf;
typedef __attribute__((ext_vector_type(8)))  float  v8f;
typedef __attribute__((ext_vector_type(4)))  unsigned int u32x4;
typedef __attribute__((ext_vector_type(8)))  int i32x8;
typedef __attribute__((ext_vector_type(4)))  int i32x4;

__device__ __forceinline__ unsigned short f2bf(float f) {
  unsigned u = __float_as_uint(f);
  unsigned r = (u + 0x7FFFu + ((u >> 16) & 1u)) >> 16;  // round-to-nearest-even
  return (unsigned short)r;
}

// ---------------------------------------------------------------------------
// TDM: DMA a 2D bf16 tile (tileRows x tileK elems, row stride strideElems)
// from global memory into LDS at lds_addr (bytes, rows packed tileK*2 apart).
// D# layout per CDNA5 ISA 8.3/8.4. amdgpu-toolchain 6-arg builtin form:
//   (uint32x4 g0, int32x8 g1, int32x4 g2, int32x4 g3, int32x8 g4, i32 cpol)
// ---------------------------------------------------------------------------
__device__ __forceinline__ void tdm_load_2d(unsigned lds_addr, const void* gp,
                                            unsigned tileK, unsigned tileRows,
                                            unsigned strideElems) {
  unsigned long long ga = (unsigned long long)(uintptr_t)gp;
  u32x4 g0;
  g0[0] = 1u;                                  // count=1 (valid user descriptor)
  g0[1] = lds_addr;                            // lds_addr[31:0]
  g0[2] = (unsigned)ga;                        // global_addr[31:0]
  g0[3] = (unsigned)((ga >> 32) & 0x01FFFFFFu) | (2u << 30);  // ga[56:32] | type=2
  const unsigned td0 = strideElems;            // tensor_dim0 (elements per line)
  const unsigned td1 = 0x00100000u;            // tensor_dim1: large (no OOB clip)
  const unsigned long long s0 = strideElems;   // tensor_dim0_stride (elements)
  i32x8 g1;
  g1[0] = 0x00010000;                          // data_size=1 -> 2-byte elements
  g1[1] = (int)((td0 & 0xFFFFu) << 16);        // [63:48] = tensor_dim0 lo16
  g1[2] = (int)(((td0 >> 16) & 0xFFFFu) | ((td1 & 0xFFFFu) << 16));
  g1[3] = (int)(((td1 >> 16) & 0xFFFFu) | (tileK << 16));   // tile_dim0
  g1[4] = (int)(tileRows & 0xFFFFu);           // tile_dim1 (tile_dim2 = 0)
  g1[5] = (int)(unsigned)(s0 & 0xFFFFFFFFu);   // dim0_stride lo32
  g1[6] = (int)(unsigned)((s0 >> 32) & 0xFFFFu);  // dim0_stride hi16 | dim1_stride lo = 0
  g1[7] = 0;
  i32x4 gz4 = {0, 0, 0, 0};                    // groups 2/3: unused (2D tile)
  i32x8 gz8 = {0, 0, 0, 0, 0, 0, 0, 0};
  __builtin_amdgcn_tensor_load_to_lds(g0, g1, gz4, gz4, gz8, 0);
}

// ---------------------------------------------------------------------------
// f32 -> bf16 conversion kernels
// ---------------------------------------------------------------------------
__global__ void cvt_f32_bf16(const float* __restrict__ src,
                             unsigned short* __restrict__ dst, long n) {
  long i = (long)blockIdx.x * 256 + threadIdx.x;
  if (i < n) dst[i] = f2bf(src[i]);
}

// dst is [dstR x dstK]; src is [srcR x srcK]; zero-pad outside.
__global__ void cvt_pad_bf16(const float* __restrict__ src,
                             unsigned short* __restrict__ dst,
                             int srcR, int srcK, int dstK, long n) {
  long i = (long)blockIdx.x * 256 + threadIdx.x;
  if (i >= n) return;
  long r = i / dstK;
  int  c = (int)(i % dstK);
  unsigned short v = 0;
  if (r < srcR && c < srcK) v = f2bf(src[r * (long)srcK + c]);
  dst[i] = v;
}

__global__ void pad_bias_f32(const float* __restrict__ src, float* __restrict__ dst,
                             int n, int np) {
  int i = threadIdx.x;
  if (i < np) dst[i] = (i < n) ? src[i] : 0.0f;
}

// ---------------------------------------------------------------------------
// Operator softmax weights + entropy + conflict (tiny; one wave)
// ---------------------------------------------------------------------------
__global__ void op_weights_kernel(const float* __restrict__ logits,  // [T,K]
                                  float* __restrict__ wts,           // [T,K]
                                  float* __restrict__ conflict,      // [T]
                                  float* __restrict__ entropy) {     // [T]
  int t = threadIdx.x;
  if (t >= TT) return;
  float l[KOP], m = -1e30f;
  for (int k = 0; k < KOP; ++k) { l[k] = logits[t * KOP + k]; m = fmaxf(m, l[k]); }
  float s = 0.f;
  for (int k = 0; k < KOP; ++k) { l[k] = __expf(l[k] - m); s += l[k]; }
  float w[KOP], sm = 0.f;
  for (int k = 0; k < KOP; ++k) { w[k] = l[k] / s; wts[t * KOP + k] = w[k]; sm += w[k]; }
  float mean = sm / KOP, var = 0.f;
  for (int k = 0; k < KOP; ++k) { float d = w[k] - mean; var += d * d; }
  conflict[t] = sqrtf(var / (KOP - 1));  // unbiased std
  float ent = 0.f;
  for (int k = 0; k < KOP; ++k) {
    float up = w[k] / (sm + 1e-6f);
    ent -= up * __logf(fmaxf(up, 1e-6f));
  }
  entropy[t] = ent;
}

// ---------------------------------------------------------------------------
// Shared epilogue (EPI codes)
//  0 = f32 store
//  1 = f32 store + bf16 store at temper-permuted row (xp -> xg)
//  2 = relu -> bf16 store
//  3 = relu, atomicAdd(outF, scale * val)   (operator mixture)
// ---------------------------------------------------------------------------
template <int EPI>
__device__ __forceinline__ void epilogue_elem(float val, int row, int col, int ldo,
                                              long oOff, float sc,
                                              float* __restrict__ outF,
                                              unsigned short* __restrict__ outB) {
  if (EPI == 2 || EPI == 3) val = fmaxf(val, 0.0f);
  const long idx = oOff + (long)row * ldo + col;
  if (EPI == 0) {
    outF[idx] = val;
  } else if (EPI == 1) {
    outF[idx] = val;
    const int pr = (row % TT) * GG + row / TT;  // xp row p -> xg row t*g+gi
    outB[(long)pr * ldo + col] = f2bf(val);
  } else if (EPI == 2) {
    outB[idx] = f2bf(val);
  } else if (EPI == 3) {
    atomicAdd(&outF[idx], sc * val);
  }
}

// ---------------------------------------------------------------------------
// TDM + LDS double-buffered WMMA GEMM:  out = EPI( A[M,K] * W[N,K]^T + bias )
// Block tile 64x256, 8 waves (2x4), wave tile 32x64, k-step 32.
// Wave 0 drives the Tensor Data Mover; everyone computes from LDS.
// Batched over blockIdx.z (see strides/div params).
// ---------------------------------------------------------------------------
template <int EPI>
__global__ __launch_bounds__(256)
void gemm_bf16_wmma_tdm(const unsigned short* __restrict__ Abase,
                        const unsigned short* __restrict__ Wbase,
                        const float* __restrict__ biasBase,
                        float* __restrict__ outF,
                        unsigned short* __restrict__ outB,
                        int Kd, int lda, int ldw, int ldo,
                        long strideA, long strideW, long strideBias, long strideO,
                        int divA, int divO,
                        const float* __restrict__ scale) {
  constexpr int MT = 2, NT = 4, WM = 2, WN = 4;
  __shared__ unsigned short As[2][64 * 32];    //  8 KB: A tiles, rows of 32 bf16
  __shared__ unsigned short Bs[2][256 * 32];   // 32 KB: B tiles, rows of 32 bf16

  const int z = blockIdx.z;
  const unsigned short* A = Abase + (long)(z / divA) * strideA;
  const unsigned short* W = Wbase + (long)z * strideW;
  const float* bias = biasBase + (long)z * strideBias;
  const long oOff = (long)(z / divO) * strideO;

  const int wave = threadIdx.x >> 5;
  const int lane = threadIdx.x & 31;
  const int wm = wave / WN, wn = wave % WN;
  const int r = lane & 15, hh = lane >> 4;

  const int blockM0 = blockIdx.x * (WM * MT * 16);  // 64-row block tile
  const int blockN0 = blockIdx.y * (WN * NT * 16);  // 256-col block tile

  const unsigned ldsA[2] = {(unsigned)(uintptr_t)(void*)&As[0][0],
                            (unsigned)(uintptr_t)(void*)&As[1][0]};
  const unsigned ldsB[2] = {(unsigned)(uintptr_t)(void*)&Bs[0][0],
                            (unsigned)(uintptr_t)(void*)&Bs[1][0]};

  v8f c[MT][NT] = {};

  // prologue: stage k0 = 0 tiles into buffer 0
  if (wave == 0) {
    tdm_load_2d(ldsA[0], A + (long)blockM0 * lda, 32, 64, (unsigned)lda);
    tdm_load_2d(ldsB[0], W + (long)blockN0 * ldw, 32, 256, (unsigned)ldw);
    __builtin_amdgcn_s_wait_tensorcnt(0);
  }
  __syncthreads();

  int buf = 0;
  for (int k0 = 0; k0 < Kd; k0 += 32) {
    // stage next k tile into the other buffer while we compute
    if (wave == 0 && (k0 + 32) < Kd) {
      tdm_load_2d(ldsA[buf ^ 1], A + (long)blockM0 * lda + (k0 + 32), 32, 64,
                  (unsigned)lda);
      tdm_load_2d(ldsB[buf ^ 1], W + (long)blockN0 * ldw + (k0 + 32), 32, 256,
                  (unsigned)ldw);
    }

    const unsigned short* as = &As[buf][0];
    const unsigned short* bs = &Bs[buf][0];
    v16bf a[MT], b[NT];
#pragma unroll
    for (int mt = 0; mt < MT; ++mt) {
      // A 16x32 bf16 layout: lane r/h holds K = 16h..16h+15 contiguous
      const int arow = wm * 32 + mt * 16 + r;
      v8bf lo = *reinterpret_cast<const v8bf*>(as + arow * 32 + 16 * hh);
      v8bf hi = *reinterpret_cast<const v8bf*>(as + arow * 32 + 16 * hh + 8);
#pragma unroll
      for (int i = 0; i < 8; ++i) { a[mt][i] = lo[i]; a[mt][i + 8] = hi[i]; }
    }
#pragma unroll
    for (int nt = 0; nt < NT; ++nt) {
      // B 32x16 layout: lane holds col N=r, K = 16h..16h+15 contiguous
      const int bcol = wn * 64 + nt * 16 + r;
      b[nt] = *reinterpret_cast<const v16bf*>(bs + bcol * 32 + 16 * hh);
    }
#pragma unroll
    for (int mt = 0; mt < MT; ++mt)
#pragma unroll
      for (int nt = 0; nt < NT; ++nt)
        c[mt][nt] = __builtin_amdgcn_wmma_f32_16x16x32_bf16(
            false, a[mt], false, b[nt], (short)0, c[mt][nt], false, false);

    if (wave == 0) __builtin_amdgcn_s_wait_tensorcnt(0);
    __syncthreads();
    buf ^= 1;
  }

  const float sc = (EPI == 3) ? scale[z] : 1.0f;
#pragma unroll
  for (int mt = 0; mt < MT; ++mt) {
#pragma unroll
    for (int nt = 0; nt < NT; ++nt) {
      const int col = blockN0 + wn * 64 + nt * 16 + r;
      const float bv = bias[col];
#pragma unroll
      for (int v = 0; v < 8; ++v) {
        const int row = blockM0 + wm * 32 + mt * 16 + v + 8 * hh;
        epilogue_elem<EPI>(c[mt][nt][v] + bv, row, col, ldo, oOff, sc, outF, outB);
      }
    }
  }
}

// ---------------------------------------------------------------------------
// Direct-global WMMA GEMM (for the small N=16 logits head).
// ---------------------------------------------------------------------------
template <int MT, int NT, int WM, int WN, int EPI>
__global__ __launch_bounds__(WM * WN * 32)
void gemm_bf16_wmma(const unsigned short* __restrict__ Abase,
                    const unsigned short* __restrict__ Wbase,
                    const float* __restrict__ biasBase,
                    float* __restrict__ outF,
                    unsigned short* __restrict__ outB,
                    int Kd, int lda, int ldw, int ldo,
                    const float* __restrict__ scale) {
  const unsigned short* A = Abase;
  const unsigned short* W = Wbase;
  const float* bias = biasBase;

  const int wave = threadIdx.x >> 5;
  const int lane = threadIdx.x & 31;
  const int wm = wave / WN, wn = wave % WN;
  const int r = lane & 15, hh = lane >> 4;

  const int tileM0 = (blockIdx.x * WM + wm) * (MT * 16);
  const int tileN0 = (blockIdx.y * WN + wn) * (NT * 16);

  v8f c[MT][NT] = {};

  for (int k0 = 0; k0 < Kd; k0 += 32) {
    v16bf a[MT], b[NT];
#pragma unroll
    for (int mt = 0; mt < MT; ++mt) {
      long rowOff = (long)(tileM0 + mt * 16 + r) * lda;
      v8bf lo = *reinterpret_cast<const v8bf*>(A + rowOff + k0 + 16 * hh);
      v8bf hi = *reinterpret_cast<const v8bf*>(A + rowOff + k0 + 16 * hh + 8);
#pragma unroll
      for (int i = 0; i < 8; ++i) { a[mt][i] = lo[i]; a[mt][i + 8] = hi[i]; }
    }
#pragma unroll
    for (int nt = 0; nt < NT; ++nt) {
      long colOff = (long)(tileN0 + nt * 16 + r) * ldw;
      b[nt] = *reinterpret_cast<const v16bf*>(W + colOff + k0 + 16 * hh);
    }
#pragma unroll
    for (int mt = 0; mt < MT; ++mt)
#pragma unroll
      for (int nt = 0; nt < NT; ++nt)
        c[mt][nt] = __builtin_amdgcn_wmma_f32_16x16x32_bf16(
            false, a[mt], false, b[nt], (short)0, c[mt][nt], false, false);
  }

  const float sc = 1.0f;
#pragma unroll
  for (int mt = 0; mt < MT; ++mt) {
#pragma unroll
    for (int nt = 0; nt < NT; ++nt) {
      const int col = tileN0 + nt * 16 + r;
      const float bv = bias[col];
#pragma unroll
      for (int v = 0; v < 8; ++v) {
        const int row = tileM0 + mt * 16 + v + 8 * hh;
        epilogue_elem<EPI>(c[mt][nt][v] + bv, row, col, ldo, 0, sc, outF, outB);
      }
    }
  }
  (void)scale;
}

// ---------------------------------------------------------------------------
// Per-row stats: one block per grouped row rp = t*g+gi
// rowstats[rp] = {pred_err, novelty, sparsity, plasticity} (means over H)
// ---------------------------------------------------------------------------
__global__ __launch_bounds__(256)
void stats_row_kernel(const float* __restrict__ og,     // [T*g, H] grouped
                      const float* __restrict__ xp,     // [B, H] original order
                      const float* __restrict__ noise,  // [B, H] original order
                      float* __restrict__ rowstats) {   // [T*g, 4]
  const int rp = blockIdx.x;
  const int t = rp / GG, gi = rp % GG;
  const long p = (long)gi * TT + t;
  const int tid = threadIdx.x;
  float s0 = 0.f, s1 = 0.f, s2 = 0.f, s3 = 0.f;
  for (int c = tid; c < HH; c += 256) {
    float xv = xp[p * HH + c];
    float ov = og[(long)rp * HH + c];
    float nv = noise[p * HH + c] * 0.01f;
    float d = ov - xv;
    s0 += d * d; s1 += xv * xv; s2 += fabsf(ov); s3 += nv * nv;
  }
  __shared__ float sh0[256], sh1[256], sh2[256], sh3[256];
  sh0[tid] = s0; sh1[tid] = s1; sh2[tid] = s2; sh3[tid] = s3;
  __syncthreads();
  for (int off = 128; off > 0; off >>= 1) {
    if (tid < off) {
      sh0[tid] += sh0[tid + off]; sh1[tid] += sh1[tid + off];
      sh2[tid] += sh2[tid + off]; sh3[tid] += sh3[tid + off];
    }
    __syncthreads();
  }
  if (tid == 0) {
    const float inv = 1.0f / HH;
    rowstats[(long)rp * 4 + 0] = sh0[0] * inv;
    rowstats[(long)rp * 4 + 1] = sh1[0] * inv;
    rowstats[(long)rp * 4 + 2] = sh2[0] * inv;
    rowstats[(long)rp * 4 + 3] = sh3[0] * inv;
  }
}

// ---------------------------------------------------------------------------
// Per-temper finalization -> signals[T,8], novelty[T], plast[T]
// ---------------------------------------------------------------------------
__global__ __launch_bounds__(256)
void stats_final_kernel(const float* __restrict__ rowstats,
                        const float* __restrict__ entropy,
                        float* __restrict__ signals,   // [T,8] (7 used)
                        float* __restrict__ novelty,   // [T]
                        float* __restrict__ plast) {   // [T]
  const int t = blockIdx.x;
  const int tid = threadIdx.x;
  float a0 = 0.f, a1 = 0.f, a2 = 0.f, a3 = 0.f;
  for (int gi = tid; gi < GG; gi += 256) {
    const float* rs = rowstats + ((long)t * GG + gi) * 4;
    a0 += rs[0]; a1 += rs[1]; a2 += rs[2]; a3 += rs[3];
  }
  __shared__ float sh0[256], sh1[256], sh2[256], sh3[256];
  __shared__ float means[4];
  sh0[tid] = a0; sh1[tid] = a1; sh2[tid] = a2; sh3[tid] = a3;
  __syncthreads();
  for (int off = 128; off > 0; off >>= 1) {
    if (tid < off) {
      sh0[tid] += sh0[tid + off]; sh1[tid] += sh1[tid + off];
      sh2[tid] += sh2[tid + off]; sh3[tid] += sh3[tid + off];
    }
    __syncthreads();
  }
  if (tid == 0) {
    means[0] = sh0[0] / GG; means[1] = sh1[0] / GG;
    means[2] = sh2[0] / GG; means[3] = sh3[0] / GG;
  }
  __syncthreads();
  const float navg = 0.01f * means[0];
  float v = 0.f;
  for (int gi = tid; gi < GG; gi += 256) {
    float pe = rowstats[((long)t * GG + gi) * 4];
    float dd = pe - navg;
    v += dd * dd;
  }
  sh0[tid] = v;
  __syncthreads();
  for (int off = 128; off > 0; off >>= 1) {
    if (tid < off) sh0[tid] += sh0[tid + off];
    __syncthreads();
  }
  if (tid == 0) {
    float pe_mean = means[0];
    signals[t * 8 + 0] = means[3];      // plasticity mean
    signals[t * 8 + 1] = means[1];      // novelty mean
    signals[t * 8 + 2] = pe_mean;       // pred_err mean
    signals[t * 8 + 3] = entropy[t];
    signals[t * 8 + 4] = means[2];      // sparsity mean
    signals[t * 8 + 5] = -pe_mean;      // reward_delta mean
    signals[t * 8 + 6] = sh0[0] / GG;   // reward_var
    novelty[t] = means[1];
    plast[t]   = pe_mean;               // plast_t == pred_err mean
  }
}

// ---------------------------------------------------------------------------
// Growth controller MLP: 7 -> H -> 2 per temper
// ---------------------------------------------------------------------------
__global__ __launch_bounds__(256)
void growth_mlp_kernel(const float* __restrict__ gW1,  // [T,H,7]
                       const float* __restrict__ gb1,  // [T,H]
                       const float* __restrict__ gW2,  // [T,2,H]
                       const float* __restrict__ gb2,  // [T,2]
                       const float* __restrict__ signals,
                       float* __restrict__ dec) {      // [T,2]
  const int t = blockIdx.x, tid = threadIdx.x;
  float s[7];
  for (int j = 0; j < 7; ++j) s[j] = signals[t * 8 + j];
  float d0 = 0.f, d1 = 0.f;
  for (int h = tid; h < HH; h += 256) {
    const float* w = gW1 + ((long)t * HH + h) * 7;
    float acc = gb1[t * HH + h];
    for (int j = 0; j < 7; ++j) acc += w[j] * s[j];
    acc = fmaxf(acc, 0.0f);
    d0 += gW2[((long)t * 2 + 0) * HH + h] * acc;
    d1 += gW2[((long)t * 2 + 1) * HH + h] * acc;
  }
  __shared__ float sh0[256], sh1[256];
  sh0[tid] = d0; sh1[tid] = d1;
  __syncthreads();
  for (int off = 128; off > 0; off >>= 1) {
    if (tid < off) { sh0[tid] += sh0[tid + off]; sh1[tid] += sh1[tid + off]; }
    __syncthreads();
  }
  if (tid == 0) {
    dec[t * 2 + 0] = sh0[0] + gb2[t * 2 + 0];
    dec[t * 2 + 1] = sh1[0] + gb2[t * 2 + 1];
  }
}

// ---------------------------------------------------------------------------
// Scatter grouped out -> original order (f32 to d_out, bf16 for pred GEMM)
// ---------------------------------------------------------------------------
__global__ __launch_bounds__(256)
void scatter_out_kernel(const float* __restrict__ og, float* __restrict__ out_full,
                        unsigned short* __restrict__ out_bf) {
  const long p = blockIdx.x;
  const int t = (int)(p % TT);
  const long gi = p / TT;
  const long rp = (long)t * GG + gi;
  for (int c = threadIdx.x; c < HH; c += 256) {
    float v = og[rp * HH + c];
    out_full[p * HH + c] = v;
    out_bf[p * HH + c] = f2bf(v);
  }
}

// ---------------------------------------------------------------------------
// Enriched router input: [out | id_emb(4) | novelty | conflict | plast | pad]
// ---------------------------------------------------------------------------
__global__ __launch_bounds__(256)
void build_enriched_kernel(const float* __restrict__ og,
                           const float* __restrict__ id_emb,   // [T,4]
                           const float* __restrict__ novelty,
                           const float* __restrict__ conflict,
                           const float* __restrict__ plast,
                           unsigned short* __restrict__ enr) { // [T*g, KP]
  const long rp = blockIdx.x;
  const int t = (int)(rp / GG);
  unsigned short* e = enr + rp * KP;
  for (int c = threadIdx.x; c < KP; c += 256) {
    float v;
    if (c < HH)            v = og[rp * HH + c];
    else if (c < HH + 4)   v = id_emb[t * 4 + (c - HH)];
    else if (c == HH + 4)  v = novelty[t];
    else if (c == HH + 5)  v = conflict[t];
    else if (c == HH + 6)  v = plast[t];
    else                   v = 0.0f;
    e[c] = f2bf(v);
  }
}

// ---------------------------------------------------------------------------
// Pack logits [T*g,16] grouped -> [B,13] original order
// ---------------------------------------------------------------------------
__global__ void pack_logits_kernel(const float* __restrict__ lg,
                                   float* __restrict__ out) {
  long i = (long)blockIdx.x * 256 + threadIdx.x;
  const long n = (long)BB * 13;
  if (i >= n) return;
  long p = i / 13;
  int  j = (int)(i % 13);
  int  t = (int)(p % TT);
  long gi = p / TT;
  out[i] = lg[((long)t * GG + gi) * NPL + j];
}

// ---------------------------------------------------------------------------
// Launch
// ---------------------------------------------------------------------------
extern "C" void kernel_launch(void* const* d_in, const int* in_sizes, int n_in,
                              void* d_out, int out_size, void* d_ws, size_t ws_size,
                              hipStream_t stream) {
  const float* x         = (const float*)d_in[0];
  const float* Wp        = (const float*)d_in[1];
  const float* bp        = (const float*)d_in[2];
  const float* opW1      = (const float*)d_in[3];
  const float* opb1      = (const float*)d_in[4];
  const float* opW2      = (const float*)d_in[5];
  const float* opb2      = (const float*)d_in[6];
  const float* op_logits = (const float*)d_in[7];
  const float* id_emb    = (const float*)d_in[8];
  const float* rW1       = (const float*)d_in[9];
  const float* rb1       = (const float*)d_in[10];
  const float* rW2       = (const float*)d_in[11];
  const float* rb2       = (const float*)d_in[12];
  const float* gW1       = (const float*)d_in[13];
  const float* gb1       = (const float*)d_in[14];
  const float* gW2       = (const float*)d_in[15];
  const float* gb2       = (const float*)d_in[16];
  const float* phW       = (const float*)d_in[17];
  const float* phb       = (const float*)d_in[18];
  const float* noise     = (const float*)d_in[19];

  float* out_full   = (float*)d_out;                       // [B,H]
  float* logits_out = out_full + (size_t)BB * HH;          // [B,13]
  float* pred_out   = logits_out + (size_t)BB * 13;        // [B,D]
  float* dec_out    = pred_out + (size_t)BB * DD;          // [T,2]

  // ---- workspace carve-up (~206 MB) ----
  char* ws = (char*)d_ws;
  size_t off = 0;
  auto alloc = [&](size_t bytes) -> void* {
    void* p = ws + off;
    off = (off + bytes + 255) & ~(size_t)255;
    return p;
  };
  unsigned short* xbf    = (unsigned short*)alloc((size_t)BB * DD * 2);
  unsigned short* Wpbf   = (unsigned short*)alloc((size_t)HH * DD * 2);
  unsigned short* opW1bf = (unsigned short*)alloc((size_t)TT * KOP * HH * HH * 2);
  unsigned short* opW2bf = (unsigned short*)alloc((size_t)TT * KOP * HH * HH * 2);
  unsigned short* phWbf  = (unsigned short*)alloc((size_t)DD * HH * 2);
  unsigned short* rW1bf  = (unsigned short*)alloc((size_t)HH * KP * 2);
  unsigned short* rW2bf  = (unsigned short*)alloc((size_t)NPL * HH * 2);
  float*          xp     = (float*)alloc((size_t)BB * HH * 4);
  unsigned short* xgbf   = (unsigned short*)alloc((size_t)BB * HH * 2);
  unsigned short* h1bf   = (unsigned short*)alloc((size_t)TT * KOP * GG * HH * 2);
  float*          og     = (float*)alloc((size_t)BB * HH * 4);
  float*          rowst  = (float*)alloc((size_t)BB * 4 * 4);
  unsigned short* outbf  = (unsigned short*)alloc((size_t)BB * HH * 2);
  unsigned short* enr    = (unsigned short*)alloc((size_t)BB * KP * 2);
  unsigned short* rhbf   = (unsigned short*)alloc((size_t)BB * HH * 2);
  float*          lgits  = (float*)alloc((size_t)BB * NPL * 4);
  float*          small  = (float*)alloc(1024 * 4);
  float* wts = small;            // 36
  float* conflict = small + 64;  // 12
  float* entropy  = small + 96;  // 12
  float* signals  = small + 128; // 96
  float* novelty  = small + 256; // 12
  float* plast    = small + 288; // 12
  float* rb2p     = small + 320; // 16

  auto cdiv = [](long a, long b) { return (unsigned)((a + b - 1) / b); };

  // ---- conversions ----
  long n;
  n = (long)BB * DD; cvt_f32_bf16<<<cdiv(n, 256), 256, 0, stream>>>(x, xbf, n);
  n = (long)HH * DD; cvt_f32_bf16<<<cdiv(n, 256), 256, 0, stream>>>(Wp, Wpbf, n);
  n = (long)TT * KOP * HH * HH;
  cvt_f32_bf16<<<cdiv(n, 256), 256, 0, stream>>>(opW1, opW1bf, n);
  cvt_f32_bf16<<<cdiv(n, 256), 256, 0, stream>>>(opW2, opW2bf, n);
  n = (long)DD * HH; cvt_f32_bf16<<<cdiv(n, 256), 256, 0, stream>>>(phW, phWbf, n);
  n = (long)HH * KP;
  cvt_pad_bf16<<<cdiv(n, 256), 256, 0, stream>>>(rW1, rW1bf, HH, HH + 7, KP, n);
  n = (long)NPL * HH;
  cvt_pad_bf16<<<cdiv(n, 256), 256, 0, stream>>>(rW2, rW2bf, TT + 1, HH, HH, n);
  pad_bias_f32<<<1, 32, 0, stream>>>(rb2, rb2p, TT + 1, NPL);

  // ---- operator mixture weights ----
  op_weights_kernel<<<1, 32, 0, stream>>>(op_logits, wts, conflict, entropy);

  // ---- zero the mixture accumulator ----
  (void)hipMemsetAsync(og, 0, (size_t)BB * HH * 4, stream);

  // ---- GEMM 1: xp = x @ Wp^T + bp  (also emit temper-grouped xg in bf16) ----
  gemm_bf16_wmma_tdm<1><<<dim3(BB / 64, HH / 256, 1), 256, 0, stream>>>(
      xbf, Wpbf, bp, xp, xgbf, DD, DD, DD, HH, 0, 0, 0, 0, 1, 1, nullptr);

  // ---- GEMM 2 (batched z = t*K+k): h1 = relu(xg_t @ opW1^T + opb1) ----
  gemm_bf16_wmma_tdm<2><<<dim3(GG / 64, HH / 256, TT * KOP), 256, 0, stream>>>(
      xgbf, opW1bf, opb1, nullptr, h1bf, HH, HH, HH, HH,
      (long)GG * HH, (long)HH * HH, HH, (long)GG * HH, KOP, 1, nullptr);

  // ---- GEMM 3 (batched): og[t] += w[t,k] * relu(h1 @ opW2^T + opb2) ----
  gemm_bf16_wmma_tdm<3><<<dim3(GG / 64, HH / 256, TT * KOP), 256, 0, stream>>>(
      h1bf, opW2bf, opb2, og, nullptr, HH, HH, HH, HH,
      (long)GG * HH, (long)HH * HH, HH, (long)GG * HH, 1, KOP, wts);

  // ---- stats, growth MLP, scatter, enriched ----
  stats_row_kernel<<<BB, 256, 0, stream>>>(og, xp, noise, rowst);
  stats_final_kernel<<<TT, 256, 0, stream>>>(rowst, entropy, signals, novelty, plast);
  growth_mlp_kernel<<<TT, 256, 0, stream>>>(gW1, gb1, gW2, gb2, signals, dec_out);
  scatter_out_kernel<<<BB, 256, 0, stream>>>(og, out_full, outbf);
  build_enriched_kernel<<<BB, 256, 0, stream>>>(og, id_emb, novelty, conflict, plast, enr);

  // ---- GEMM 4: rh = relu(enriched @ rW1p^T + rb1) ----
  gemm_bf16_wmma_tdm<2><<<dim3(BB / 64, HH / 256, 1), 256, 0, stream>>>(
      enr, rW1bf, rb1, nullptr, rhbf, KP, KP, KP, HH, 0, 0, 0, 0, 1, 1, nullptr);

  // ---- GEMM 5: logits = rh @ rW2p^T + rb2p  (N=16 tile, direct-global) ----
  gemm_bf16_wmma<4, 1, 8, 1, 0><<<dim3(BB / 512, 1, 1), 256, 0, stream>>>(
      rhbf, rW2bf, rb2p, lgits, nullptr, HH, HH, HH, NPL, nullptr);

  pack_logits_kernel<<<cdiv((long)BB * 13, 256), 256, 0, stream>>>(lgits, logits_out);

  // ---- GEMM 6: pred = out_full @ phW^T + phb ----
  gemm_bf16_wmma_tdm<0><<<dim3(BB / 64, DD / 256, 1), 256, 0, stream>>>(
      outbf, phWbf, phb, pred_out, nullptr, HH, HH, HH, DD, 0, 0, 0, 0, 1, 1, nullptr);
}
